// HeteroRGCNClassifier_11269994184928
// MI455X (gfx1250) — matrix operand order repaired and checked by reading, not compiled, not run on hardware
//
#include <hip/hip_runtime.h>
#include <hip/hip_bf16.h>

typedef __attribute__((ext_vector_type(16))) _Float16 v16h;
typedef __attribute__((ext_vector_type(8)))  float    v8f;

#define N_NODES 50000
#define N_EDGES 625000
#define HID     128
#define STRIPS  (N_NODES / 16)        // 3125, N_NODES % 16 == 0
#define WPREP   (8 * 4 * 32 * 16)     // halfs per prepped 128x128 weight matrix

// ---------------- atomics ----------------
__device__ __forceinline__ void atomAddF(float* p, float v) {
  // hardware global_atomic_add_f32 path
  unsafeAtomicAdd(p, v);
}

// ---------------- utility kernels ----------------
__global__ void zero_f32(float* __restrict__ p, long n) {
  long i = (long)blockIdx.x * blockDim.x + threadIdx.x;
  long stride = (long)gridDim.x * blockDim.x;
  for (; i < n; i += stride) p[i] = 0.0f;
}

// h0 = relu(x @ emb_w + emb_b), x:[N,3], emb_w:[3,128]
__global__ void embed_kernel(const float* __restrict__ x, const float* __restrict__ w,
                             const float* __restrict__ b, float* __restrict__ h) {
  int idx = blockIdx.x * blockDim.x + threadIdx.x;
  if (idx >= N_NODES * HID) return;
  int n = idx >> 7, c = idx & 127;
  float v = fmaf(x[n * 3 + 0], w[c],
            fmaf(x[n * 3 + 1], w[HID + c],
            fmaf(x[n * 3 + 2], w[2 * HID + c], b[c])));
  h[idx] = fmaxf(v, 0.0f);
}

__global__ void deg_kernel(const int* __restrict__ dst, float* __restrict__ deg, int E) {
  int i = blockIdx.x * blockDim.x + threadIdx.x;
  if (i < E) atomAddF(&deg[dst[i]], 1.0f);
}

// in-place deg -> 1/max(deg,1)
__global__ void rdeg_kernel(float* __restrict__ degF, float* __restrict__ degS, int n) {
  int i = blockIdx.x * blockDim.x + threadIdx.x;
  if (i >= n) return;
  degF[i] = 1.0f / fmaxf(degF[i], 1.0f);
  degS[i] = 1.0f / fmaxf(degS[i], 1.0f);
}

// one wave per edge: gather 128 floats from h[src], fp32-atomic-scatter to agg[dst]
__global__ void scatter_kernel(const float* __restrict__ h, const int* __restrict__ src,
                               const int* __restrict__ dst, float* __restrict__ agg, int E) {
  int wid  = blockIdx.x * (blockDim.x >> 5) + (threadIdx.x >> 5);
  int lane = threadIdx.x & 31;
  if (wid >= E) return;
  int s = src[wid], d = dst[wid];
  float4 v = ((const float4*)(h + (long)s * HID))[lane];
  float* o = agg + (long)d * HID + lane * 4;
  atomAddF(o + 0, v.x);
  atomAddF(o + 1, v.y);
  atomAddF(o + 2, v.z);
  atomAddF(o + 3, v.w);
}

// Pre-swizzle a 128x128 fp32 weight matrix (optionally sum of two) into the
// per-lane WMMA B-fragment layout, converted to f16.
// B fragment (16x16x32 f16): lane l (n = l%16, g = l/16), half j -> B[8*g + (j&7) + 16*(j>>3)][n]
// out[(((t*4+q)*32)+l)*16 + j] for col-tile t, k-chunk q.
__global__ void prep_w_kernel(const float* __restrict__ W0, const float* __restrict__ W1,
                              _Float16* __restrict__ out) {
  int idx = blockIdx.x * blockDim.x + threadIdx.x;
  if (idx >= WPREP) return;
  int j = idx & 15;
  int l = (idx >> 4) & 31;
  int q = (idx >> 9) & 3;
  int t = idx >> 11;
  int g = l >> 4, n = l & 15;
  int k = 8 * g + (j & 7) + 16 * (j >> 3);
  float v = W0[(q * 32 + k) * HID + t * 16 + n];
  if (W1) v += W1[(q * 32 + k) * HID + t * 16 + n];
  out[idx] = (_Float16)v;
}

__global__ void addvec_kernel(const float* __restrict__ a, const float* __restrict__ b,
                              float* __restrict__ o, int n) {
  int i = blockIdx.x * blockDim.x + threadIdx.x;
  if (i < n) o[i] = a[i] + b[i];
}

// ---------------- WMMA layer kernel ----------------
// A fragment (16x32 f16): lane l (m = l%16, g = l/16): halves 0..7 -> k = 8g..8g+7,
// halves 8..15 -> k = 16+8g..16+8g+7 (ISA 16-bit A table). Scale fused in (deg norm).
__device__ __forceinline__ v16h load_a_frag(const float* __restrict__ rowp, int q, int g,
                                            float scale) {
  const float4* p0 = (const float4*)(rowp + q * 32 + 8 * g);
  const float4* p1 = (const float4*)(rowp + q * 32 + 16 + 8 * g);
  float4 r0 = p0[0], r1 = p0[1];
  float4 r2 = p1[0], r3 = p1[1];
  v16h a;
  a[0]  = (_Float16)(r0.x * scale); a[1]  = (_Float16)(r0.y * scale);
  a[2]  = (_Float16)(r0.z * scale); a[3]  = (_Float16)(r0.w * scale);
  a[4]  = (_Float16)(r1.x * scale); a[5]  = (_Float16)(r1.y * scale);
  a[6]  = (_Float16)(r1.z * scale); a[7]  = (_Float16)(r1.w * scale);
  a[8]  = (_Float16)(r2.x * scale); a[9]  = (_Float16)(r2.y * scale);
  a[10] = (_Float16)(r2.z * scale); a[11] = (_Float16)(r2.w * scale);
  a[12] = (_Float16)(r3.x * scale); a[13] = (_Float16)(r3.y * scale);
  a[14] = (_Float16)(r3.z * scale); a[15] = (_Float16)(r3.w * scale);
  return a;
}

// out = op( 0.5 * ( (aggF/degF) @ BrelF + (aggS/degS) @ BrelS + h @ (BrootF+BrootS) + bias ) )
// One wave per 16-row strip; 8 col tiles x (3 matrices x 4 k-chunks) WMMAs.
// Accumulator starts at inline 0 (WMMA SRC2 = inline constant); bias folded into epilogue.
__global__ void __launch_bounds__(256) layer_kernel(
    const float* __restrict__ h, const float* __restrict__ aggF, const float* __restrict__ aggS,
    const float* __restrict__ rdegF, const float* __restrict__ rdegS,
    const _Float16* __restrict__ BrelF, const _Float16* __restrict__ BrelS,
    const _Float16* __restrict__ Bsum, const float* __restrict__ bias,
    float* __restrict__ out, int doRelu) {
  int strip = blockIdx.x * (blockDim.x >> 5) + (threadIdx.x >> 5);
  if (strip >= STRIPS) return;                 // wave-uniform: EXEC stays all-ones for WMMA
  int lane = threadIdx.x & 31;
  int g = lane >> 4, m = lane & 15;
  long row = (long)strip * 16 + m;
  float sF = rdegF[row];
  float sS = rdegS[row];
  const float* hp = h    + row * HID;
  const float* fp = aggF + row * HID;
  const float* sp = aggS + row * HID;

  v16h aH[4], aF[4], aS[4];
#pragma unroll
  for (int q = 0; q < 4; ++q) {
    aH[q] = load_a_frag(hp, q, g, 1.0f);
    aF[q] = load_a_frag(fp, q, g, sF);
    aS[q] = load_a_frag(sp, q, g, sS);
  }

  long outRow0 = (long)strip * 16 + 8 * g;     // C/D layout: VGPR v -> row 8g+v, col = lane%16
#pragma unroll
  for (int t = 0; t < 8; ++t) {
    v8f c = {};                                // inline-0 C operand for first WMMA
#pragma unroll
    for (int q = 0; q < 4; ++q) {
      long boff = (long)((t * 4 + q) * 32 + lane) * 16;
      // hoist all three fragment loads: one 6x b128 clause, WMMAs drain the counter
      v16h bF = *(const v16h*)(BrelF + boff);
      v16h bS = *(const v16h*)(BrelS + boff);
      v16h bH = *(const v16h*)(Bsum + boff);
      c = __builtin_amdgcn_wmma_f32_16x16x32_f16(false, aF[q], false, bF, (short)0, c, false, false);
      c = __builtin_amdgcn_wmma_f32_16x16x32_f16(false, aS[q], false, bS, (short)0, c, false, false);
      c = __builtin_amdgcn_wmma_f32_16x16x32_f16(false, aH[q], false, bH, (short)0, c, false, false);
    }
    float bv = 0.5f * bias[t * 16 + m];
#pragma unroll
    for (int v = 0; v < 8; ++v) {
      float x = fmaf(c[v], 0.5f, bv);
      if (doRelu) x = fmaxf(x, 0.0f);
      out[(outRow0 + v) * HID + t * 16 + m] = x;
    }
  }
}

// logits = h @ out_w + out_b, out_w:[128,6]
__global__ void out_kernel(const float* __restrict__ h, const float* __restrict__ w,
                           const float* __restrict__ b, float* __restrict__ out) {
  int idx = blockIdx.x * blockDim.x + threadIdx.x;
  if (idx >= N_NODES * 6) return;
  int n = idx / 6, c = idx % 6;
  const float* hp = h + (long)n * HID;
  float acc = b[c];
#pragma unroll
  for (int k = 0; k < HID; k += 4) {
    float4 hv = *(const float4*)(hp + k);
    acc = fmaf(hv.x, w[(k + 0) * 6 + c], acc);
    acc = fmaf(hv.y, w[(k + 1) * 6 + c], acc);
    acc = fmaf(hv.z, w[(k + 2) * 6 + c], acc);
    acc = fmaf(hv.w, w[(k + 3) * 6 + c], acc);
  }
  out[idx] = acc;
}

// ---------------- host orchestration ----------------
extern "C" void kernel_launch(void* const* d_in, const int* in_sizes, int n_in,
                              void* d_out, int out_size, void* d_ws, size_t ws_size,
                              hipStream_t stream) {
  const float* x       = (const float*)d_in[0];
  const int*   eiF     = (const int*)d_in[1];
  const int*   eiS     = (const int*)d_in[2];
  const float* emb_w   = (const float*)d_in[3];
  const float* emb_b   = (const float*)d_in[4];
  const float* wrel0f  = (const float*)d_in[5];
  const float* wroot0f = (const float*)d_in[6];
  const float* b0f     = (const float*)d_in[7];
  const float* wrel0s  = (const float*)d_in[8];
  const float* wroot0s = (const float*)d_in[9];
  const float* b0s     = (const float*)d_in[10];
  const float* wrel1f  = (const float*)d_in[11];
  const float* wroot1f = (const float*)d_in[12];
  const float* b1f     = (const float*)d_in[13];
  const float* wrel1s  = (const float*)d_in[14];
  const float* wroot1s = (const float*)d_in[15];
  const float* b1s     = (const float*)d_in[16];
  const float* out_w   = (const float*)d_in[17];
  const float* out_b   = (const float*)d_in[18];

  const int* srcF = eiF;            // ei shape (2, E): row 0 = src
  const int* dstF = eiF + N_EDGES;  //                  row 1 = dst
  const int* srcS = eiS;
  const int* dstS = eiS + N_EDGES;

  const long NH = (long)N_NODES * HID;
  float* ws    = (float*)d_ws;
  float* hA    = ws;                //  N x 128
  float* hB    = hA + NH;           //  N x 128
  float* aggF  = hB + NH;           //  N x 128
  float* aggS  = aggF + NH;         //  N x 128
  float* degF  = aggS + NH;         //  N  (becomes 1/max(deg,1))
  float* degS  = degF + N_NODES;    //  N
  float* bsum0 = degS + N_NODES;    //  128
  float* bsum1 = bsum0 + HID;       //  128
  _Float16* Bp = (_Float16*)(bsum1 + HID);
  _Float16* BrelF0 = Bp + 0L * WPREP;
  _Float16* BrelS0 = Bp + 1L * WPREP;
  _Float16* Bsum0  = Bp + 2L * WPREP;
  _Float16* BrelF1 = Bp + 3L * WPREP;
  _Float16* BrelS1 = Bp + 4L * WPREP;
  _Float16* Bsum1  = Bp + 5L * WPREP;

  // embedding
  embed_kernel<<<(N_NODES * HID + 255) / 256, 256, 0, stream>>>(x, emb_w, emb_b, hA);

  // degrees (shared across layers) -> reciprocal
  zero_f32<<<256, 256, 0, stream>>>(degF, 2L * N_NODES);
  deg_kernel<<<(N_EDGES + 255) / 256, 256, 0, stream>>>(dstF, degF, N_EDGES);
  deg_kernel<<<(N_EDGES + 255) / 256, 256, 0, stream>>>(dstS, degS, N_EDGES);
  rdeg_kernel<<<(N_NODES + 255) / 256, 256, 0, stream>>>(degF, degS, N_NODES);

  // pre-swizzled f16 weights; root weights summed across relations
  prep_w_kernel<<<WPREP / 256, 256, 0, stream>>>(wrel0f, nullptr, BrelF0);
  prep_w_kernel<<<WPREP / 256, 256, 0, stream>>>(wrel0s, nullptr, BrelS0);
  prep_w_kernel<<<WPREP / 256, 256, 0, stream>>>(wroot0f, wroot0s, Bsum0);
  prep_w_kernel<<<WPREP / 256, 256, 0, stream>>>(wrel1f, nullptr, BrelF1);
  prep_w_kernel<<<WPREP / 256, 256, 0, stream>>>(wrel1s, nullptr, BrelS1);
  prep_w_kernel<<<WPREP / 256, 256, 0, stream>>>(wroot1f, wroot1s, Bsum1);
  addvec_kernel<<<1, HID, 0, stream>>>(b0f, b0s, bsum0, HID);
  addvec_kernel<<<1, HID, 0, stream>>>(b1f, b1s, bsum1, HID);

  const int scatterBlocks = (N_EDGES + 7) / 8;          // 8 waves (edges) per 256-thread block
  const int layerBlocks   = (STRIPS + 7) / 8;

  // ---- layer 0 ----
  zero_f32<<<4096, 256, 0, stream>>>(aggF, 2L * NH);    // aggF and aggS are contiguous
  scatter_kernel<<<scatterBlocks, 256, 0, stream>>>(hA, srcF, dstF, aggF, N_EDGES);
  scatter_kernel<<<scatterBlocks, 256, 0, stream>>>(hA, srcS, dstS, aggS, N_EDGES);
  layer_kernel<<<layerBlocks, 256, 0, stream>>>(hA, aggF, aggS, degF, degS,
                                                BrelF0, BrelS0, Bsum0, bsum0, hB, 1);

  // ---- layer 1 (no relu) ----
  zero_f32<<<4096, 256, 0, stream>>>(aggF, 2L * NH);
  scatter_kernel<<<scatterBlocks, 256, 0, stream>>>(hB, srcF, dstF, aggF, N_EDGES);
  scatter_kernel<<<scatterBlocks, 256, 0, stream>>>(hB, srcS, dstS, aggS, N_EDGES);
  layer_kernel<<<layerBlocks, 256, 0, stream>>>(hB, aggF, aggS, degF, degS,
                                                BrelF1, BrelS1, Bsum1, bsum1, hA, 0);

  // ---- head ----
  out_kernel<<<(N_NODES * 6 + 255) / 256, 256, 0, stream>>>(hA, out_w, out_b, (float*)d_out);
}
